// ModuleBARO_39281770889660
// MI455X (gfx1250) — matrix operand order repaired and checked by reading, not compiled, not run on hardware
//
#include <hip/hip_runtime.h>
#include <hip/hip_bf16.h>
#include <stdint.h>

typedef __bf16 bf16;
typedef __attribute__((ext_vector_type(16))) __bf16 v16bf;
typedef __attribute__((ext_vector_type(8)))  __bf16 v8bf;
typedef __attribute__((ext_vector_type(8)))  float  v8f;

// Problem dims
#define DT 16
#define DB 32
#define DN 512
#define DF 128
#define DE 2048
#define MROWS (DB*DN)   // 16384

// GEMM tiling
#define BM 128
#define BN 128
#define BK 64

// ---------------------------------------------------------------------------
// Stage a BM x BK A-tile global -> LDS with gfx1250 async-to-LDS DMA.
// Contiguous 16B chunks on both sides; tracked with ASYNCcnt (no VGPR data).
// ---------------------------------------------------------------------------
__device__ __forceinline__
void stage_A_async(const bf16* __restrict__ A, long long lda,
                   int m0, int k0, bf16* dst, int tid)
{
    #pragma unroll
    for (int it = 0; it < (BM*BK)/(256*8); ++it) {
        int e = (it*256 + tid) * 8;
        int r = e >> 6;            // /BK
        int c = e & (BK-1);
        unsigned long long g =
            (unsigned long long)(A + (long long)(m0 + r)*lda + k0 + c);
        // generic LDS pointer: low 32 bits are the LDS byte offset
        unsigned l = (unsigned)(unsigned long long)(dst + r*BK + c);
        asm volatile("global_load_async_to_lds_b128 %0, %1, off"
                     :: "v"(l), "v"(g) : "memory");
    }
}

// Load the BK x BN B-tile into registers (all loads in flight together).
__device__ __forceinline__
void load_B_regs(const bf16* __restrict__ Bm, long long ldb,
                 int n0, int k0, int tid, v8bf breg[4])
{
    #pragma unroll
    for (int it = 0; it < (BN*BK)/(256*8); ++it) {
        int e  = (it*256 + tid) * 8;
        int kk = e >> 7;           // /BN
        int nn = e & (BN-1);
        breg[it] = *(const v8bf*)(Bm + (long long)(k0 + kk)*ldb + n0 + nn);
    }
}

// Scatter B registers into LDS as Bs[n][k] (K contiguous per row).
__device__ __forceinline__
void scatter_B(bf16* dst, int tid, const v8bf breg[4])
{
    #pragma unroll
    for (int it = 0; it < (BN*BK)/(256*8); ++it) {
        int e  = (it*256 + tid) * 8;
        int kk = e >> 7;
        int nn = e & (BN-1);
        #pragma unroll
        for (int q = 0; q < 8; ++q) dst[(nn + q)*BK + kk] = breg[it][q];
    }
}

// 2x4 WMMA 16x16x32 over one staged tile (two k-steps of 32).
__device__ __forceinline__
void compute_tile(const bf16* AsB, const bf16* BsB,
                  int row0, int col0, int lrow, int hi, int base0,
                  v8f acc[2][4])
{
    #pragma unroll
    for (int kk = 0; kk < BK; kk += 32) {
        union AU { v16bf v; v8bf h[2]; } a[2];
        #pragma unroll
        for (int i = 0; i < 2; ++i) {
            const bf16* ap = AsB + (row0 + i*16 + lrow)*BK + kk;
            a[i].h[0] = *(const v8bf*)(ap + base0);        // K = base0..+7
            a[i].h[1] = *(const v8bf*)(ap + base0 + 8);    // K = base0+8..+15
        }
        v16bf b[4];
        #pragma unroll
        for (int j = 0; j < 4; ++j)
            b[j] = *(const v16bf*)(BsB + (col0 + j*16 + lrow)*BK + kk + hi*16);

        #pragma unroll
        for (int i = 0; i < 2; ++i)
            #pragma unroll
            for (int j = 0; j < 4; ++j)
                acc[i][j] = __builtin_amdgcn_wmma_f32_16x16x32_bf16(
                    false, a[i].v, false, b[j], (short)0, acc[i][j],
                    false, false);
    }
}

// ---------------------------------------------------------------------------
// Generic bf16 WMMA GEMM: C = A[M,K] * B[K,N] (+bias[N]); optional fp32 out,
// bf16 out, and transposed-bf16 out. grid.z batches via element strides.
// Block = 256 threads = 8 waves; wave computes 32x64; LDS double-buffered
// with async global->LDS staging overlapping the WMMA stream.
// ---------------------------------------------------------------------------
__global__ __launch_bounds__(256)
void gemm_wmma_bf16(const bf16* __restrict__ A, long long lda, long long sA,
                    const bf16* __restrict__ Bm, long long ldb, long long sB,
                    const float* __restrict__ bias,
                    float* __restrict__ Cf, long long ldcf,
                    bf16*  __restrict__ Cb, long long ldcb,
                    bf16*  __restrict__ Ct, long long ldct,
                    long long sC, int M, int N, int K)
{
    __shared__ alignas(64) bf16 As[2][BM * BK];   // [m][k]
    __shared__ alignas(64) bf16 Bs[2][BN * BK];   // [n][k]

    const int tid  = threadIdx.x;
    const int lane = tid & 31;
    const int w    = tid >> 5;       // wave 0..7
    const int row0 = (w >> 1) * 32;  // wave tile origin in block tile
    const int col0 = (w & 1) * 64;
    const int lrow = lane & 15;      // M row (A) / N col (B,C) within 16-tile
    const int hi   = lane >> 4;      // half-wave select
    const int base0 = hi * 16;       // A-fragment K base (contiguous, matches B)

    const long long z = blockIdx.z;
    A  += z * sA;
    Bm += z * sB;
    if (Cf) Cf += z * sC;
    if (Cb) Cb += z * sC;

    const int m0 = blockIdx.y * BM;
    const int n0 = blockIdx.x * BN;

    v8f acc[2][4];
    const v8f vzero = {0.f,0.f,0.f,0.f,0.f,0.f,0.f,0.f};
    #pragma unroll
    for (int i = 0; i < 2; ++i)
        #pragma unroll
        for (int j = 0; j < 4; ++j) acc[i][j] = vzero;

    const int nk = K / BK;

    // ---- prologue: stage tile 0 into buffer 0
    {
        stage_A_async(A, lda, m0, 0, As[0], tid);
        v8bf b0[4];
        load_B_regs(Bm, ldb, n0, 0, tid, b0);
        scatter_B(Bs[0], tid, b0);
        asm volatile("s_wait_asynccnt 0x0" ::: "memory");
        __syncthreads();
    }

    int buf = 0;
    for (int kt = 0; kt < nk; ++kt) {
        const int nbuf = buf ^ 1;
        const bool have_next = (kt + 1) < nk;

        v8bf bnext[4];
        if (have_next) {
            const int k1 = (kt + 1) * BK;
            stage_A_async(A, lda, m0, k1, As[nbuf], tid);   // async, no VGPRs
            load_B_regs(Bm, ldb, n0, k1, tid, bnext);       // in flight under WMMA
        }

        compute_tile(As[buf], Bs[buf], row0, col0, lrow, hi, base0, acc);

        if (have_next) scatter_B(Bs[nbuf], tid, bnext);

        asm volatile("s_wait_asynccnt 0x0" ::: "memory");
        __syncthreads();
        buf = nbuf;
    }

    // ---- epilogue: C/D layout VGPR r -> M = r + 8*hi, N = lane&15
    #pragma unroll
    for (int i = 0; i < 2; ++i) {
        #pragma unroll
        for (int j = 0; j < 4; ++j) {
            const int gn = n0 + col0 + j*16 + lrow;
            const float bia = bias ? bias[gn] : 0.0f;
            #pragma unroll
            for (int r = 0; r < 8; ++r) {
                const int gm = m0 + row0 + i*16 + hi*8 + r;
                const float val = acc[i][j][r] + bia;
                if (Cf) Cf[(long long)gm*ldcf + gn] = val;
                if (Cb) Cb[(long long)gm*ldcb + gn] = (bf16)val;
                if (Ct) Ct[(long long)gn*ldct + gm] = (bf16)val;
            }
        }
    }
}

// x[T,B,N,F] -> Xb[(b*N+n), f*T+t] bf16  (the permute(1,2,3,0).reshape)
__global__ __launch_bounds__(256)
void pack_x_kernel(const float* __restrict__ x, bf16* __restrict__ Xb)
{
    int idx = blockIdx.x * 256 + threadIdx.x;   // B*N*E = 33,554,432
    int m = idx >> 11;          // row index (b*N+n)
    int e = idx & 2047;
    int b = m >> 9, n = m & 511;
    int f = e >> 4, t = e & 15;
    Xb[idx] = (bf16)x[(((t*DB + b)*DN + n) << 7) + f];
}

// Wt[k,n] = (bf16)W[n,k]   (2048x2048)
__global__ __launch_bounds__(256)
void packT_w_kernel(const float* __restrict__ W, bf16* __restrict__ Wt)
{
    int idx = blockIdx.x * 256 + threadIdx.x;   // 4,194,304
    int k = idx >> 11, n = idx & 2047;
    Wt[idx] = (bf16)W[n*2048 + k];
}

// Row softmax of energy/sqrt(E), in-place fp32 + bf16 copy for P@V
__global__ __launch_bounds__(256)
void softmax_kernel(float* __restrict__ attn, bf16* __restrict__ Pb)
{
    const float scale = 0.022097086912079608f;  // 1/sqrt(2048)
    const long long row = blockIdx.x;           // 0 .. B*N-1
    float* e = attn + row * DN;
    bf16*  p = Pb   + row * DN;
    const int tid = threadIdx.x;

    __shared__ float red[256];
    float v0 = e[tid]       * scale;
    float v1 = e[tid + 256] * scale;

    red[tid] = fmaxf(v0, v1);
    __syncthreads();
    for (int s = 128; s > 0; s >>= 1) {
        if (tid < s) red[tid] = fmaxf(red[tid], red[tid + s]);
        __syncthreads();
    }
    const float mx = red[0];
    __syncthreads();

    float p0 = __expf(v0 - mx), p1 = __expf(v1 - mx);
    red[tid] = p0 + p1;
    __syncthreads();
    for (int s = 128; s > 0; s >>= 1) {
        if (tid < s) red[tid] += red[tid + s];
        __syncthreads();
    }
    const float inv = 1.0f / red[0];
    p0 *= inv; p1 *= inv;
    e[tid] = p0;       e[tid + 256] = p1;
    p[tid] = (bf16)p0; p[tid + 256] = (bf16)p1;
}

// out1[t,n,f] = mean_b x[t,b,n,f] + mean_b' OutFC_view[n,b',t,f]
// OutFC_view flat index = n*65536 + b'*2048 + t*128 + f  (the "buggy" reshape)
__global__ __launch_bounds__(256)
void combine_kernel(const float* __restrict__ x, const float* __restrict__ OutFC,
                    float* __restrict__ out1)
{
    int idx = blockIdx.x * 256 + threadIdx.x;   // T*N*F = 1,048,576
    int t = idx >> 16;
    int rem = idx & 65535;
    int n = rem >> 7, f = rem & 127;
    const float* xp = x     + (long long)t * 2097152 + n*128 + f;  // stride b: N*F
    const float* op = OutFC + (long long)n * 65536   + t*128 + f;  // stride b': T*F
    float sx = 0.f, so = 0.f;
    #pragma unroll
    for (int b = 0; b < 32; ++b) { sx += xp[b * 65536]; so += op[b * 2048]; }
    out1[idx] = (sx + so) * (1.0f / 32.0f);
}

extern "C" void kernel_launch(void* const* d_in, const int* in_sizes, int n_in,
                              void* d_out, int out_size, void* d_ws, size_t ws_size,
                              hipStream_t stream)
{
    const float* x   = (const float*)d_in[0];
    const float* Wq  = (const float*)d_in[1];
    const float* bq  = (const float*)d_in[2];
    const float* Wk  = (const float*)d_in[3];
    const float* bk  = (const float*)d_in[4];
    const float* Wv  = (const float*)d_in[5];
    const float* bv  = (const float*)d_in[6];
    const float* Wfc = (const float*)d_in[7];
    const float* bfc = (const float*)d_in[8];

    float* out1 = (float*)d_out;                     // [T,N,F] = 1,048,576 f32
    float* attn = out1 + (size_t)DT * DN * DF;       // [B,1,N,N] = 8,388,608 f32

    // -------- workspace layout (256B-aligned slices) --------
    char* ws = (char*)d_ws;
    size_t off = 0;
    auto alloc = [&](size_t bytes) -> void* {
        void* p = ws + off;
        off += (bytes + 255) & ~(size_t)255;
        return p;
    };
    bf16* Xb = (bf16*)alloc((size_t)MROWS * DE * 2);     // 64 MB packed input
    bf16* Wt = (bf16*)alloc((size_t)DE * DE * 2);        // 8 MB, reused per weight
    bf16* Qb = (bf16*)alloc((size_t)MROWS * DE * 2);     // 64 MB
    bf16* Kt = (bf16*)alloc((size_t)DE * MROWS * 2);     // 64 MB (K transposed)
    bf16* Vb = (bf16*)alloc((size_t)MROWS * DE * 2);     // 64 MB
    bf16* Pb = (bf16*)alloc((size_t)DB * DN * DN * 2);   // 16 MB attention bf16
    bf16*  Out2  = Qb;                                   // reuse Q after energy
    float* OutFC = (float*)Kt;                           // reuse Kt+Vb (128 MB)

    // -------- 1. pack activations --------
    pack_x_kernel<<<(MROWS * DE) / 256, 256, 0, stream>>>(x, Xb);

    const dim3 gBig(DE / BN, MROWS / BM, 1);   // (16,128,1)
    const dim3 gEng(DN / BN, DN / BM, DB);     // (4,4,32)
    const dim3 gPV (DE / BN, DN / BM, DB);     // (16,4,32)

    // -------- 2. Q = Xb @ Wq^T + bq  (bf16 out) --------
    packT_w_kernel<<<(DE * DE) / 256, 256, 0, stream>>>(Wq, Wt);
    gemm_wmma_bf16<<<gBig, 256, 0, stream>>>(Xb, DE, 0, Wt, DE, 0, bq,
                                             nullptr, 0, Qb, DE, nullptr, 0,
                                             0, MROWS, DE, DE);
    // -------- 3. K = Xb @ Wk^T + bk  (bf16, stored transposed -> Kt[d, m]) ---
    packT_w_kernel<<<(DE * DE) / 256, 256, 0, stream>>>(Wk, Wt);
    gemm_wmma_bf16<<<gBig, 256, 0, stream>>>(Xb, DE, 0, Wt, DE, 0, bk,
                                             nullptr, 0, nullptr, 0, Kt, MROWS,
                                             0, MROWS, DE, DE);
    // -------- 4. V = Xb @ Wv^T + bv  (bf16 out) --------
    packT_w_kernel<<<(DE * DE) / 256, 256, 0, stream>>>(Wv, Wt);
    gemm_wmma_bf16<<<gBig, 256, 0, stream>>>(Xb, DE, 0, Wt, DE, 0, bv,
                                             nullptr, 0, Vb, DE, nullptr, 0,
                                             0, MROWS, DE, DE);
    // -------- 5. energy[b] = Q[b] @ K[b]^T  (fp32, into d_out attn region) ---
    gemm_wmma_bf16<<<gEng, 256, 0, stream>>>(Qb, DE, (long long)DN * DE,
                                             Kt, MROWS, (long long)DN,
                                             nullptr,
                                             attn, DN, nullptr, 0, nullptr, 0,
                                             (long long)DN * DN, DN, DN, DE);
    // -------- 6. softmax(energy/sqrt(E)) in place + bf16 copy --------
    softmax_kernel<<<DB * DN, 256, 0, stream>>>(attn, Pb);
    // -------- 7. Out2[b] = P[b] @ V[b]  (bf16 out, reuses Q buffer) --------
    gemm_wmma_bf16<<<gPV, 256, 0, stream>>>(Pb, DN, (long long)DN * DN,
                                            Vb, DE, (long long)DN * DE,
                                            nullptr,
                                            nullptr, 0, Out2, DE, nullptr, 0,
                                            (long long)DN * DE, DN, DE, DN);
    // -------- 8. OutFC = Out2 @ Wfc^T + bfc  (fp32, reuses Kt+Vb region) ----
    packT_w_kernel<<<(DE * DE) / 256, 256, 0, stream>>>(Wfc, Wt);
    gemm_wmma_bf16<<<gBig, 256, 0, stream>>>(Out2, DE, 0, Wt, DE, 0, bfc,
                                             OutFC, DE, nullptr, 0, nullptr, 0,
                                             0, MROWS, DE, DE);
    // -------- 9. final: x.mean(batch) + buggy-view mean --------
    combine_kernel<<<(DT * DN * DF) / 256, 256, 0, stream>>>(x, OutFC, out1);
}
